// MLDecoder_26903675142149
// MI455X (gfx1250) — matrix operand order repaired and checked by reading, not compiled, not run on hardware
//
#include <hip/hip_runtime.h>
#include <hip/hip_bf16.h>
#include <cstddef>
#include <cstdint>

typedef __attribute__((ext_vector_type(16))) _Float16 v16h;
typedef __attribute__((ext_vector_type(8)))  _Float16 v8h;
typedef __attribute__((ext_vector_type(8)))  float    v8f;

// ---------------------------------------------------------------------------
// CDNA5 async global->LDS staging via inline asm (portable across toolchains;
// bypasses the clang builtin whose pointer-arg types differ between ROCm
// releases).  vdst = LDS byte offset VGPR, vaddr = 64-bit global address.
// ---------------------------------------------------------------------------
#define USE_ASYNC_LDS 1

__device__ __forceinline__ void async_copy_b128(const void* gsrc, void* lds_dst) {
    unsigned loff = (unsigned)(unsigned long long)
        (__attribute__((address_space(3))) char*)lds_dst;   // LDS byte offset
    asm volatile("global_load_async_to_lds_b128 %0, %1, off"
                 :: "v"(loff), "v"(gsrc) : "memory");
}

__device__ __forceinline__ void async_wait0() {
    asm volatile("s_wait_asynccnt 0x0" ::: "memory");
}

// ---------------------------------------------------------------------------
// Generic f32 -> f16 conversion
// ---------------------------------------------------------------------------
__global__ void f2h_kernel(const float* __restrict__ in, _Float16* __restrict__ out,
                           size_t n) {
    size_t i = (size_t)blockIdx.x * blockDim.x + threadIdx.x;
    if (i < n) out[i] = (_Float16)in[i];
}

// ---------------------------------------------------------------------------
// Weight convert + transpose: in (K x N, f32, row-major) -> out (N x K, f16)
// One-time pass so the GEMM can stage B tiles with contiguous-K vector loads.
// ---------------------------------------------------------------------------
__global__ void f2h_t_kernel(const float* __restrict__ in, _Float16* __restrict__ out,
                             int K, int N) {
    size_t i = (size_t)blockIdx.x * blockDim.x + threadIdx.x;
    size_t total = (size_t)K * N;
    if (i >= total) return;
    int k = (int)(i % K);
    int n = (int)(i / K);
    out[i] = (_Float16)in[(size_t)k * N + n];
}

// ---------------------------------------------------------------------------
// x (B, F, 7, 7) -> Xh (B*49, F) f16   (transpose channel/spatial + convert)
// ---------------------------------------------------------------------------
__global__ void xpose_kernel(const float* __restrict__ x, _Float16* __restrict__ Xh) {
    const size_t total = (size_t)128 * 49 * 2048;
    size_t i = (size_t)blockIdx.x * blockDim.x + threadIdx.x;
    if (i >= total) return;
    int f = (int)(i % 2048);
    size_t rem = i / 2048;
    int s = (int)(rem % 49);
    int b = (int)(rem / 49);
    Xh[i] = (_Float16)x[((size_t)b * 2048 + f) * 49 + s];
}

// ---------------------------------------------------------------------------
// Tiled WMMA GEMM:  C(M,N) = op( A(M,K) @ Bt(N,K)^T + bias )
// A: f16 row-major [M][K].  Bt: f16 PRE-TRANSPOSED [N][K].  C: f32.
// block = 256 threads (8 waves), tile 128x64x32; 32 WMMAs per K-step.
// K % 32 == 0, N % 64 == 0 guaranteed by call sites; M handled by row clamp
// (out-of-range rows compute garbage that is never stored).
// ---------------------------------------------------------------------------
#define GTM 128
#define GTN 64
#define GTK 32

__global__ __launch_bounds__(256) void gemm_wmma_f16(
    const _Float16* __restrict__ A, const _Float16* __restrict__ Bt,
    const float* __restrict__ bias, float* __restrict__ C,
    int M, int N, int Kdim, int relu)
{
    __shared__ alignas(16) _Float16 As[GTM][GTK];   // [m][k]
    __shared__ alignas(16) _Float16 Bs[GTN][GTK];   // [n][k]

    const int tid  = threadIdx.x;       // 0..255
    const int wave = tid >> 5;          // 0..7
    const int lane = tid & 31;          // wave32
    const int m0 = blockIdx.y * GTM;
    const int n0 = blockIdx.x * GTN;

    // Staging coordinates: 8-half (16B) chunks, 4 chunks per 32-half row.
    const int am   = tid >> 2;              // A rows 0..63 (j=0) / +64 (j=1)
    const int koff = (tid & 3) * 8;         // chunk offset in K
    const int bn   = tid >> 2;              // B row (n) 0..63
    // Row-clamped global row indices (guard only matters when M % GTM != 0)
    const int gmA0 = min(m0 + am,      M - 1);
    const int gmA1 = min(m0 + am + 64, M - 1);
    const int gnB  = n0 + bn;

    v8f acc0 = {}, acc1 = {}, acc2 = {}, acc3 = {};

    const int mrow  = wave * 16 + (lane & 15);
    const int khalf = (lane >> 4) * 8;      // A fragment K base (0 or 8)
    const int kb    = (lane >> 4) * 8;      // B fragment K base (0 or 8)

    for (int k0 = 0; k0 < Kdim; k0 += GTK) {
#ifdef USE_ASYNC_LDS
        async_copy_b128(A  + (size_t)gmA0 * Kdim + (k0 + koff), &As[am][koff]);
        async_copy_b128(A  + (size_t)gmA1 * Kdim + (k0 + koff), &As[am + 64][koff]);
        async_copy_b128(Bt + (size_t)gnB  * Kdim + (k0 + koff), &Bs[bn][koff]);
        async_wait0();
#else
        *(v8h*)&As[am][koff]      = *(const v8h*)(A  + (size_t)gmA0 * Kdim + (k0 + koff));
        *(v8h*)&As[am + 64][koff] = *(const v8h*)(A  + (size_t)gmA1 * Kdim + (k0 + koff));
        *(v8h*)&Bs[bn][koff]      = *(const v8h*)(Bt + (size_t)gnB  * Kdim + (k0 + koff));
#endif
        __syncthreads();

        // A fragment: lane L holds M=L%16, K = {khalf+0..7} U {16+khalf+0..7}
        v8h alo = *(const v8h*)&As[mrow][khalf];
        v8h ahi = *(const v8h*)&As[mrow][16 + khalf];
        v16h afrag = __builtin_shufflevector(alo, ahi,
            0,1,2,3,4,5,6,7,8,9,10,11,12,13,14,15);

        // 4 column tiles: lane L holds N=L%16, K = {kb+0..7} U {16+kb+0..7}
        #pragma unroll
        for (int c = 0; c < 4; ++c) {
            int ncol = c * 16 + (lane & 15);
            v8h blo = *(const v8h*)&Bs[ncol][kb];
            v8h bhi = *(const v8h*)&Bs[ncol][kb + 16];
            v16h bfrag = __builtin_shufflevector(blo, bhi,
                0,1,2,3,4,5,6,7,8,9,10,11,12,13,14,15);
            v8f* pacc = (c == 0) ? &acc0 : (c == 1) ? &acc1 : (c == 2) ? &acc2 : &acc3;
            *pacc = __builtin_amdgcn_wmma_f32_16x16x32_f16(
                false, afrag, false, bfrag, (short)0, *pacc, false, false);
        }
        __syncthreads();
    }

    // Store: VGPR r <-> M = wave*16 + (lane>>4)*8 + r ; N = c*16 + lane%16
    #pragma unroll
    for (int c = 0; c < 4; ++c) {
        const v8f* pacc = (c == 0) ? &acc0 : (c == 1) ? &acc1 : (c == 2) ? &acc2 : &acc3;
        #pragma unroll
        for (int r = 0; r < 8; ++r) {
            int gm = m0 + wave * 16 + (lane >> 4) * 8 + r;
            int gn = n0 + c * 16 + (lane & 15);
            if (gm < M) {
                float v = (*pacc)[r] + bias[gn];
                if (relu) v = fmaxf(v, 0.f);
                C[(size_t)gm * N + gn] = v;
            }
        }
    }
}

// ---------------------------------------------------------------------------
// LayerNorm over D=768:  val = alpha*X[row] + (R ? R[row % rmod or row] : 0)
// ---------------------------------------------------------------------------
__global__ __launch_bounds__(256) void ln_kernel(
    const float* __restrict__ X, const float* __restrict__ R, int rmod,
    float alpha, const float* __restrict__ g, const float* __restrict__ be,
    float* __restrict__ outF, _Float16* __restrict__ outH)
{
    const int D = 768;
    const int row = blockIdx.x;
    const int rr = (rmod > 0) ? (row % rmod) : row;
    const int tid = threadIdx.x;
    __shared__ float red[256];

    float v[3];
    float s = 0.f;
    #pragma unroll
    for (int j = 0; j < 3; ++j) {
        int d = tid + 256 * j;
        float val = alpha * X[(size_t)row * D + d];
        if (R) val += R[(size_t)rr * D + d];
        v[j] = val; s += val;
    }
    red[tid] = s; __syncthreads();
    for (int off = 128; off > 0; off >>= 1) {
        if (tid < off) red[tid] += red[tid + off];
        __syncthreads();
    }
    float mean = red[0] / D;
    __syncthreads();

    float vs = 0.f;
    #pragma unroll
    for (int j = 0; j < 3; ++j) { float t = v[j] - mean; vs += t * t; }
    red[tid] = vs; __syncthreads();
    for (int off = 128; off > 0; off >>= 1) {
        if (tid < off) red[tid] += red[tid + off];
        __syncthreads();
    }
    float inv = rsqrtf(red[0] / D + 1e-5f);

    #pragma unroll
    for (int j = 0; j < 3; ++j) {
        int d = tid + 256 * j;
        float o = (v[j] - mean) * inv * g[d] + be[d];
        outF[(size_t)row * D + d] = o;
        if (outH) outH[(size_t)row * D + d] = (_Float16)o;
    }
}

// ---------------------------------------------------------------------------
// Cross attention, one (b, h) per block.  q is batch-invariant (100 x 768).
// ---------------------------------------------------------------------------
__global__ __launch_bounds__(256) void attn_kernel(
    const float* __restrict__ q, const float* __restrict__ kmat,
    const float* __restrict__ vmat, float* __restrict__ ctx)
{
    const int S = 49, KQ = 100, HD = 96, D = 768;
    const int b = blockIdx.x, h = blockIdx.y;
    const int tid = threadIdx.x;

    __shared__ float ks[49][96];
    __shared__ float vs[49][96];
    __shared__ float sc[100][52];

    for (int i = tid; i < S * HD; i += 256) {
        int s = i / HD, d = i % HD;
        size_t src = (size_t)(b * S + s) * D + h * HD + d;
        ks[s][d] = kmat[src];
        vs[s][d] = vmat[src];
    }
    __syncthreads();

    const float scale = rsqrtf(96.f);
    for (int i = tid; i < KQ * S; i += 256) {
        int kq = i / S, s = i % S;
        const float* qr = q + (size_t)kq * D + h * HD;
        float acc = 0.f;
        for (int d = 0; d < HD; ++d) acc += qr[d] * ks[s][d];
        sc[kq][s] = acc * scale;
    }
    __syncthreads();

    if (tid < KQ) {
        float mx = -1e30f;
        for (int s = 0; s < S; ++s) mx = fmaxf(mx, sc[tid][s]);
        float sum = 0.f;
        for (int s = 0; s < S; ++s) { float e = __expf(sc[tid][s] - mx); sc[tid][s] = e; sum += e; }
        float inv = 1.f / sum;
        for (int s = 0; s < S; ++s) sc[tid][s] *= inv;
    }
    __syncthreads();

    for (int i = tid; i < KQ * HD; i += 256) {
        int kq = i / HD, d = i % HD;
        float acc = 0.f;
        for (int s = 0; s < S; ++s) acc += sc[kq][s] * vs[s][d];
        ctx[(size_t)(b * KQ + kq) * D + h * HD + d] = acc;
    }
}

// ---------------------------------------------------------------------------
// GroupFC + bias: out[b, kq*10+g] = h[b,kq,:] . dup_pool[kq,:,g] + dup_bias
// ---------------------------------------------------------------------------
__global__ __launch_bounds__(320) void groupfc_kernel(
    const float* __restrict__ hmat, const float* __restrict__ pool,
    const float* __restrict__ dbias, float* __restrict__ out)
{
    const int D = 768, G = 10, KQ = 100;
    const int b = blockIdx.x, kq = blockIdx.y;
    const int g = threadIdx.x >> 5;
    const int lane = threadIdx.x & 31;
    const float* hr = hmat + (size_t)(b * KQ + kq) * D;
    const float* pr = pool + (size_t)kq * D * G;
    float acc = 0.f;
    for (int d = lane; d < D; d += 32) acc += hr[d] * pr[(size_t)d * G + g];
    for (int off = 16; off > 0; off >>= 1) acc += __shfl_xor(acc, off, 32);
    if (lane == 0) {
        int idx = kq * G + g;   // < 1000 == num_classes
        out[(size_t)b * 1000 + idx] = acc + dbias[idx];
    }
}

// ---------------------------------------------------------------------------
// Host orchestration
// ---------------------------------------------------------------------------
extern "C" void kernel_launch(void* const* d_in, const int* in_sizes, int n_in,
                              void* d_out, int out_size, void* d_ws, size_t ws_size,
                              hipStream_t stream) {
    (void)in_sizes; (void)n_in; (void)out_size; (void)ws_size;
    const int B = 128, S = 49, F = 2048, D = 768, KQ = 100, FFd = 2048;
    const size_t BS = (size_t)B * S;          // 6272
    const size_t BK = (size_t)B * KQ;         // 12800

    const float* x        = (const float*)d_in[0];
    const float* W_embed  = (const float*)d_in[1];
    const float* b_embed  = (const float*)d_in[2];
    const float* query    = (const float*)d_in[3];
    const float* wq       = (const float*)d_in[4];
    const float* wk       = (const float*)d_in[5];
    const float* wv       = (const float*)d_in[6];
    const float* bq       = (const float*)d_in[7];
    const float* bk       = (const float*)d_in[8];
    const float* bv       = (const float*)d_in[9];
    const float* wo       = (const float*)d_in[10];
    const float* bo       = (const float*)d_in[11];
    const float* w1       = (const float*)d_in[12];
    const float* b1       = (const float*)d_in[13];
    const float* w2       = (const float*)d_in[14];
    const float* b2       = (const float*)d_in[15];
    const float* g1       = (const float*)d_in[16];
    const float* be1      = (const float*)d_in[17];
    const float* g2       = (const float*)d_in[18];
    const float* be2      = (const float*)d_in[19];
    const float* g3       = (const float*)d_in[20];
    const float* be3      = (const float*)d_in[21];
    const float* dup_pool = (const float*)d_in[22];
    const float* dup_bias = (const float*)d_in[23];
    float* out = (float*)d_out;

    // Bump allocator over d_ws
    char* p = (char*)d_ws;
    auto alloc = [&](size_t bytes) -> void* {
        void* r = (void*)p;
        p += (bytes + 255) & ~(size_t)255;
        return r;
    };
    _Float16* Xh    = (_Float16*)alloc(BS * F * 2);
    _Float16* Weht  = (_Float16*)alloc((size_t)F * D * 2);     // [D][F]
    _Float16* wqht  = (_Float16*)alloc((size_t)D * D * 2);     // [D][D]
    _Float16* wkht  = (_Float16*)alloc((size_t)D * D * 2);
    _Float16* wvht  = (_Float16*)alloc((size_t)D * D * 2);
    _Float16* woht  = (_Float16*)alloc((size_t)D * D * 2);
    _Float16* w1ht  = (_Float16*)alloc((size_t)D * FFd * 2);   // [FF][D]
    _Float16* w2ht  = (_Float16*)alloc((size_t)FFd * D * 2);   // [D][FF]
    float*    mem   = (float*)   alloc(BS * D * 4);
    _Float16* memh  = (_Float16*)alloc(BS * D * 2);
    float*    kmat  = (float*)   alloc(BS * D * 4);
    float*    vmat  = (float*)   alloc(BS * D * 4);
    float*    tgt   = (float*)   alloc((size_t)KQ * D * 4);
    _Float16* tgth  = (_Float16*)alloc((size_t)KQ * D * 2);
    float*    qmat  = (float*)   alloc((size_t)KQ * D * 4);
    float*    ctx   = (float*)   alloc(BK * D * 4);
    _Float16* ctxh  = (_Float16*)alloc(BK * D * 2);
    float*    aout  = (float*)   alloc(BK * D * 4);
    float*    tgt2  = (float*)   alloc(BK * D * 4);
    _Float16* tgt2h = (_Float16*)alloc(BK * D * 2);
    float*    ff1   = (float*)   alloc(BK * FFd * 4);
    _Float16* ff1h  = (_Float16*)alloc(BK * FFd * 2);
    float*    ff2   = (float*)   alloc(BK * D * 4);
    float*    hbuf  = (float*)   alloc(BK * D * 4);

    auto cvt = [&](const float* src, _Float16* dst, size_t n) {
        f2h_kernel<<<(unsigned)((n + 255) / 256), 256, 0, stream>>>(src, dst, n);
    };
    auto cvtT = [&](const float* src, _Float16* dst, int K, int N) {
        size_t n = (size_t)K * N;
        f2h_t_kernel<<<(unsigned)((n + 255) / 256), 256, 0, stream>>>(src, dst, K, N);
    };

    // Weight convert + transpose (one-time)
    cvtT(W_embed, Weht, F, D);
    cvtT(wq, wqht, D, D);
    cvtT(wk, wkht, D, D);
    cvtT(wv, wvht, D, D);
    cvtT(wo, woht, D, D);
    cvtT(w1, w1ht, D, FFd);
    cvtT(w2, w2ht, FFd, D);

    // x transpose + convert
    {
        size_t n = BS * F;
        xpose_kernel<<<(unsigned)((n + 255) / 256), 256, 0, stream>>>(x, Xh);
    }

    // mem = relu(Xh @ W_embed + b_embed)    (6272 x 768, K=2048)
    gemm_wmma_f16<<<dim3(D / GTN, (unsigned)(BS / GTM)), 256, 0, stream>>>(
        Xh, Weht, b_embed, mem, (int)BS, D, F, 1);
    cvt(mem, memh, BS * D);

    // k / v projections (6272 x 768, K=768)
    gemm_wmma_f16<<<dim3(D / GTN, (unsigned)(BS / GTM)), 256, 0, stream>>>(
        memh, wkht, bk, kmat, (int)BS, D, D, 0);
    gemm_wmma_f16<<<dim3(D / GTN, (unsigned)(BS / GTM)), 256, 0, stream>>>(
        memh, wvht, bv, vmat, (int)BS, D, D, 0);

    // tgt = LN(2 * query_embed)  (batch-invariant, 100 rows)
    ln_kernel<<<KQ, 256, 0, stream>>>(query, nullptr, 0, 2.0f, g1, be1, tgt, tgth);

    // q = tgt @ wq + bq   (100 x 768, M row-clamped)
    gemm_wmma_f16<<<dim3(D / GTN, (KQ + GTM - 1) / GTM), 256, 0, stream>>>(
        tgth, wqht, bq, qmat, KQ, D, D, 0);

    // attention -> ctx (12800 x 768)
    attn_kernel<<<dim3(B, 8), 256, 0, stream>>>(qmat, kmat, vmat, ctx);
    cvt(ctx, ctxh, BK * D);

    // attn_out = ctx @ wo + bo   (12800 x 768, K=768)
    gemm_wmma_f16<<<dim3(D / GTN, (unsigned)(BK / GTM)), 256, 0, stream>>>(
        ctxh, woht, bo, aout, (int)BK, D, D, 0);

    // tgt2 = LN(tgt[broadcast] + attn_out)
    ln_kernel<<<(unsigned)BK, 256, 0, stream>>>(aout, tgt, KQ, 1.0f, g2, be2, tgt2, tgt2h);

    // ff1 = relu(tgt2 @ w1 + b1)   (12800 x 2048, K=768)
    gemm_wmma_f16<<<dim3(FFd / GTN, (unsigned)(BK / GTM)), 256, 0, stream>>>(
        tgt2h, w1ht, b1, ff1, (int)BK, FFd, D, 1);
    cvt(ff1, ff1h, BK * FFd);

    // ff2 = ff1 @ w2 + b2   (12800 x 768, K=2048)
    gemm_wmma_f16<<<dim3(D / GTN, (unsigned)(BK / GTM)), 256, 0, stream>>>(
        ff1h, w2ht, b2, ff2, (int)BK, D, FFd, 0);

    // h = LN(tgt2 + ff2)
    ln_kernel<<<(unsigned)BK, 256, 0, stream>>>(ff2, tgt2, 0, 1.0f, g3, be3, hbuf, nullptr);

    // logits
    groupfc_kernel<<<dim3(B, KQ), 320, 0, stream>>>(hbuf, dup_pool, dup_bias, out);
}
